// Block_5342939316243
// MI455X (gfx1250) — compile-verified, matrix operand order checked by reading, not run on hardware
//
#include <hip/hip_runtime.h>
#include <hip/hip_bf16.h>
#include <math.h>

// ---------------------------------------------------------------------------
// Types / constants
// ---------------------------------------------------------------------------
typedef __attribute__((ext_vector_type(16))) __bf16 v16bf;
typedef __attribute__((ext_vector_type(8)))  float  v8f;

constexpr int Bb   = 4;
constexpr int Nn   = 1370;
constexpr int Cc   = 1024;
constexpr int Hh   = 16;
constexpr int DHd  = 64;
constexpr int HIDh = 4096;
constexpr int Mtok = Bb * Nn;          // 5480 token rows
constexpr float QSCALE = 0.125f;       // DH^-0.5 = 64^-0.5

__device__ inline v8f vzero8() {
    v8f r;
#pragma unroll
    for (int i = 0; i < 8; ++i) r[i] = 0.0f;
    return r;
}
__device__ inline v16bf vzero16() {
    v16bf r;
#pragma unroll
    for (int i = 0; i < 16; ++i) r[i] = (__bf16)0.0f;
    return r;
}
__device__ inline v8f wmma_bf16(v16bf a, v16bf b, v8f c) {
    // D = A(16x32 bf16) * B(32x16 bf16) + C(16x16 f32)
    return __builtin_amdgcn_wmma_f32_16x16x32_bf16(
        false, a, false, b, (short)0, c, false, false);
}

// LDS byte offset of a generic pointer to a __shared__ object (AS3 ptrtoint).
typedef __attribute__((address_space(3))) const void* lds_cptr_t;
__device__ inline unsigned lds_off(const void* p) {
    return (unsigned)(unsigned long long)(lds_cptr_t)p;
}
// CDNA5 async copy: global memory -> LDS, 16 bytes per lane, ASYNCcnt-tracked.
__device__ inline void async_copy_b128(unsigned ldsoff, const void* gptr) {
    asm volatile("global_load_async_to_lds_b128 %0, %1, off"
                 :: "v"(ldsoff), "v"((unsigned long long)(size_t)gptr)
                 : "memory");
}
__device__ inline void wait_async0() {
    asm volatile("s_wait_asynccnt 0x0" ::: "memory");
}

// ---------------------------------------------------------------------------
// f32 -> bf16 cast (weights)
// ---------------------------------------------------------------------------
__global__ __launch_bounds__(256) void cast_f32_bf16(const float* __restrict__ s,
                                                     __bf16* __restrict__ d, int n) {
    int i = blockIdx.x * blockDim.x + threadIdx.x;
    int stride = gridDim.x * blockDim.x;
    for (; i < n; i += stride) d[i] = (__bf16)s[i];
}

// ---------------------------------------------------------------------------
// LayerNorm (one block per token row), f32 in -> bf16 out
// ---------------------------------------------------------------------------
__global__ __launch_bounds__(256) void layernorm_bf16(const float* __restrict__ x,
                                                      const float* __restrict__ g,
                                                      const float* __restrict__ bta,
                                                      __bf16* __restrict__ out) {
    __shared__ float s1[256];
    __shared__ float s2[256];
    const int row = blockIdx.x;
    const int tid = threadIdx.x;
    const float4 xv = reinterpret_cast<const float4*>(x + (size_t)row * Cc)[tid];
    float sum = xv.x + xv.y + xv.z + xv.w;
    float sq  = xv.x * xv.x + xv.y * xv.y + xv.z * xv.z + xv.w * xv.w;
    s1[tid] = sum; s2[tid] = sq;
    __syncthreads();
    for (int off = 128; off > 0; off >>= 1) {
        if (tid < off) { s1[tid] += s1[tid + off]; s2[tid] += s2[tid + off]; }
        __syncthreads();
    }
    const float mu  = s1[0] * (1.0f / Cc);
    const float var = s2[0] * (1.0f / Cc) - mu * mu;
    const float rs  = rsqrtf(var + 1e-5f);
    const float4 gv = reinterpret_cast<const float4*>(g)[tid];
    const float4 bv = reinterpret_cast<const float4*>(bta)[tid];
    __bf16* o = out + (size_t)row * Cc + tid * 4;
    o[0] = (__bf16)((xv.x - mu) * rs * gv.x + bv.x);
    o[1] = (__bf16)((xv.y - mu) * rs * gv.y + bv.y);
    o[2] = (__bf16)((xv.z - mu) * rs * gv.z + bv.z);
    o[3] = (__bf16)((xv.w - mu) * rs * gv.w + bv.w);
}

// ---------------------------------------------------------------------------
// Tiled WMMA GEMM:  O[M,Ncols] = A[M,K](bf16) @ W[Ncols,K]^T(bf16) + epilogue
// 256 threads = 8 waves, block tile 128x128, K-step 64, double-buffered LDS
// filled with global_load_async_to_lds_b128 (ASYNCcnt).  Out-of-range M rows
// are CLAMPED (their outputs are never stored), so no zero-fill is needed.
// Wave grid 4(M) x 2(N): each wave -> 32x64 via 2x4 fragments, 16 wmma/step.
// MODE 0: QKV split epilogue -> q(scaled)/k/v bf16 [B,H,N,DH]
// MODE 1: outf[row,col] = resid[row,col] + (acc+bias)*gamma  (f32)
// MODE 2: outb[row,col] = gelu_exact(acc+bias)               (bf16)
// ---------------------------------------------------------------------------
template <int MODE>
__global__ __launch_bounds__(256) void gemm_bf16_wmma(
    const __bf16* __restrict__ A, const __bf16* __restrict__ W,
    const float* __restrict__ bias,
    const float* __restrict__ resid, const float* __restrict__ gamma,
    float* __restrict__ outf, __bf16* __restrict__ outb,
    __bf16* __restrict__ qb, __bf16* __restrict__ kb, __bf16* __restrict__ vb,
    int M, int K, int Ncols) {
    // 64 data cols + 16 pad: 160B row stride keeps ds_load_b128 32B-aligned
    // and spreads banks.  2 buffers x (128x80x2B) = 40KB per operand.
    __shared__ __bf16 Asl[2][128][80];
    __shared__ __bf16 Wsl[2][128][80];

    const int tid     = threadIdx.x;
    const int wave    = tid >> 5;
    const int lane    = tid & 31;
    const int hl      = lane >> 4;
    const int l15     = lane & 15;
    const int waveRow = wave & 3;     // 0..3 -> 32-row slices
    const int waveCol = wave >> 2;    // 0..1 -> 64-col slices
    const int blockM  = blockIdx.x * 128;
    const int blockN  = blockIdx.y * 128;

    v8f acc[2][4];
#pragma unroll
    for (int i = 0; i < 2; ++i)
#pragma unroll
        for (int j = 0; j < 4; ++j) acc[i][j] = vzero8();

    // staging: thread t -> row t>>1, 32-elem half (t&1), 4 async b128 per tile
    const int srow = tid >> 1;
    const int scol = (tid & 1) * 32;
    const int gra  = (blockM + srow < M) ? (blockM + srow) : (M - 1);  // clamp
    const __bf16* gA = A + (size_t)gra * K + scol;
    const __bf16* gW = W + (size_t)(blockN + srow) * K + scol;

    auto stage = [&](int buf, int k0) {
#pragma unroll
        for (int i = 0; i < 4; ++i) {
            async_copy_b128(lds_off(&Asl[buf][srow][scol + 8 * i]), gA + k0 + 8 * i);
            async_copy_b128(lds_off(&Wsl[buf][srow][scol + 8 * i]), gW + k0 + 8 * i);
        }
    };

    const int nIter = K >> 6;          // K / 64 (K is 1024 or 4096)
    stage(0, 0);
    for (int it = 0; it < nIter; ++it) {
        const int cur = it & 1;
        wait_async0();                 // own async copies landed in LDS
        __syncthreads();               // everyone's copies visible; prev reads done
        if (it + 1 < nIter) stage(cur ^ 1, (it + 1) * 64);

#pragma unroll
        for (int ks = 0; ks < 2; ++ks) {
            v16bf af[2], bfr[4];
#pragma unroll
            for (int fm = 0; fm < 2; ++fm)
                af[fm] = *reinterpret_cast<v16bf*>(
                    &Asl[cur][waveRow * 32 + fm * 16 + l15][ks * 32 + hl * 16]);
#pragma unroll
            for (int fn = 0; fn < 4; ++fn)
                bfr[fn] = *reinterpret_cast<v16bf*>(
                    &Wsl[cur][waveCol * 64 + fn * 16 + l15][ks * 32 + hl * 16]);
#pragma unroll
            for (int fm = 0; fm < 2; ++fm)
#pragma unroll
                for (int fn = 0; fn < 4; ++fn)
                    acc[fm][fn] = wmma_bf16(af[fm], bfr[fn], acc[fm][fn]);
        }
    }

    // Epilogue.  C/D layout: lane 0-15 -> N=l15, M=j; lane 16-31 -> M=8+j.
#pragma unroll
    for (int fm = 0; fm < 2; ++fm) {
#pragma unroll
        for (int fn = 0; fn < 4; ++fn) {
#pragma unroll
            for (int j = 0; j < 8; ++j) {
                const int row = blockM + waveRow * 32 + fm * 16 + hl * 8 + j;
                const int col = blockN + waveCol * 64 + fn * 16 + l15;
                if (row >= M) continue;
                const float val = acc[fm][fn][j] + bias[col];
                if (MODE == 0) {
                    const int part = col >> 10;          // col / C
                    const int rem  = col & 1023;
                    const int head = rem >> 6;
                    const int d    = rem & 63;
                    const int bi   = row / Nn;
                    const int n    = row - bi * Nn;
                    const size_t dst =
                        (((size_t)(bi * Hh + head)) * Nn + n) * DHd + d;
                    if (part == 0)      qb[dst] = (__bf16)(val * QSCALE);
                    else if (part == 1) kb[dst] = (__bf16)val;
                    else                vb[dst] = (__bf16)val;
                } else if (MODE == 1) {
                    const size_t idx = (size_t)row * Ncols + col;
                    outf[idx] = resid[idx] + val * gamma[col];
                } else { // MODE 2: exact GELU
                    const float gl = 0.5f * val * (1.0f + erff(val * 0.70710678118654752f));
                    outb[(size_t)row * Ncols + col] = (__bf16)gl;
                }
            }
        }
    }
}

// ---------------------------------------------------------------------------
// Flash attention: one block (128 threads = 4 waves) per (b,h, 64 query rows).
// q is pre-scaled by DH^-0.5.  o written as [B, N, H*DH] bf16 rows (= proj input).
// ---------------------------------------------------------------------------
__global__ __launch_bounds__(128) void attn_flash(const __bf16* __restrict__ qp,
                                                  const __bf16* __restrict__ kp,
                                                  const __bf16* __restrict__ vp,
                                                  __bf16* __restrict__ op) {
    __shared__ __bf16 Klds[32][80];       // [key][dh]
    __shared__ __bf16 Vt[64][48];         // [dh][key] (transposed at staging)
    __shared__ __bf16 Plds[4][16][48];    // per-wave P relayout scratch

    const int tid  = threadIdx.x;
    const int wave = tid >> 5;
    const int lane = tid & 31;
    const int hl   = lane >> 4;
    const int l15  = lane & 15;

    const int bh   = blockIdx.y;          // b*H + h
    const int bi   = bh >> 4;
    const int hh   = bh & 15;
    const size_t base = (size_t)bh * Nn * DHd;
    const int q0   = blockIdx.x * 64 + wave * 16;

    // Q fragments (A layout): lane half selects dh 0-15 / 16-31 of the K-step.
    v16bf aq[2];
    aq[0] = vzero16(); aq[1] = vzero16();
    const int qr = q0 + l15;
    if (qr < Nn) {
        aq[0] = *reinterpret_cast<const v16bf*>(qp + base + (size_t)qr * DHd + hl * 16);
        aq[1] = *reinterpret_cast<const v16bf*>(qp + base + (size_t)qr * DHd + 32 + hl * 16);
    }

    v8f oacc[4];
#pragma unroll
    for (int i = 0; i < 4; ++i) oacc[i] = vzero8();
    float mrow[8], lrow[8];
#pragma unroll
    for (int j = 0; j < 8; ++j) { mrow[j] = -1e30f; lrow[j] = 0.0f; }

    const int keyr = tid >> 2;       // 0..31
    const int ck   = tid & 3;        // 16-elem chunk of the 64-wide dh row

    for (int key0 = 0; key0 < Nn; key0 += 32) {
        // clamp key row: invalid keys are masked in the softmax (P underflows to 0)
        const int kr = (key0 + keyr < Nn) ? (key0 + keyr) : (Nn - 1);
        const v16bf kv = *reinterpret_cast<const v16bf*>(kp + base + (size_t)kr * DHd + ck * 16);
        const v16bf vv = *reinterpret_cast<const v16bf*>(vp + base + (size_t)kr * DHd + ck * 16);
        __syncthreads();
        *reinterpret_cast<v16bf*>(&Klds[keyr][ck * 16]) = kv;
#pragma unroll
        for (int j = 0; j < 16; ++j) Vt[ck * 16 + j][keyr] = vv[j];
        __syncthreads();

        // S = Q @ K^T  (two 16x16 key tiles, K-dim 64 = 2 wmma each)
        v8f s[2];
#pragma unroll
        for (int tn = 0; tn < 2; ++tn) {
            const v16bf b0 = *reinterpret_cast<v16bf*>(&Klds[tn * 16 + l15][hl * 16]);
            const v16bf b1 = *reinterpret_cast<v16bf*>(&Klds[tn * 16 + l15][32 + hl * 16]);
            s[tn] = wmma_bf16(aq[0], b0, vzero8());
            s[tn] = wmma_bf16(aq[1], b1, s[tn]);
        }
        // mask invalid keys
#pragma unroll
        for (int tn = 0; tn < 2; ++tn) {
            if (key0 + tn * 16 + l15 >= Nn) {
#pragma unroll
                for (int j = 0; j < 8; ++j) s[tn][j] = -1e30f;
            }
        }
        // row max across the 16 lanes of each half
        float rmax[8];
#pragma unroll
        for (int j = 0; j < 8; ++j) rmax[j] = fmaxf(s[0][j], s[1][j]);
#pragma unroll
        for (int msk = 1; msk <= 8; msk <<= 1)
#pragma unroll
            for (int j = 0; j < 8; ++j)
                rmax[j] = fmaxf(rmax[j], __shfl_xor(rmax[j], msk, 32));

        float alpha[8];
#pragma unroll
        for (int j = 0; j < 8; ++j) {
            const float nm = fmaxf(mrow[j], rmax[j]);
            alpha[j] = __expf(mrow[j] - nm);
            mrow[j]  = nm;
        }
#pragma unroll
        for (int tn = 0; tn < 2; ++tn)
#pragma unroll
            for (int j = 0; j < 8; ++j) s[tn][j] = __expf(s[tn][j] - mrow[j]);

        float rsum[8];
#pragma unroll
        for (int j = 0; j < 8; ++j) rsum[j] = s[0][j] + s[1][j];
#pragma unroll
        for (int msk = 1; msk <= 8; msk <<= 1)
#pragma unroll
            for (int j = 0; j < 8; ++j) rsum[j] += __shfl_xor(rsum[j], msk, 32);
#pragma unroll
        for (int j = 0; j < 8; ++j) lrow[j] = lrow[j] * alpha[j] + rsum[j];

        // P relayout (same-wave LDS, DS ops are in-order within a wave)
#pragma unroll
        for (int tn = 0; tn < 2; ++tn)
#pragma unroll
            for (int j = 0; j < 8; ++j)
                Plds[wave][hl * 8 + j][tn * 16 + l15] = (__bf16)s[tn][j];
        const v16bf ap = *reinterpret_cast<v16bf*>(&Plds[wave][l15][hl * 16]);

#pragma unroll
        for (int fd = 0; fd < 4; ++fd) {
#pragma unroll
            for (int j = 0; j < 8; ++j) oacc[fd][j] *= alpha[j];
            const v16bf bv = *reinterpret_cast<v16bf*>(&Vt[fd * 16 + l15][hl * 16]);
            oacc[fd] = wmma_bf16(ap, bv, oacc[fd]);
        }
    }

    // store O / l : [B, N, H*DH] bf16
#pragma unroll
    for (int fd = 0; fd < 4; ++fd) {
#pragma unroll
        for (int j = 0; j < 8; ++j) {
            const int ro = q0 + hl * 8 + j;
            if (ro < Nn)
                op[((size_t)(bi * Nn + ro)) * Cc + hh * DHd + fd * 16 + l15] =
                    (__bf16)(oacc[fd][j] / lrow[j]);
        }
    }
}

// ---------------------------------------------------------------------------
// Launch
// ---------------------------------------------------------------------------
extern "C" void kernel_launch(void* const* d_in, const int* in_sizes, int n_in,
                              void* d_out, int out_size, void* d_ws, size_t ws_size,
                              hipStream_t stream) {
    (void)in_sizes; (void)n_in; (void)out_size; (void)ws_size;
    const float* x      = (const float*)d_in[0];
    const float* ln1_g  = (const float*)d_in[1];
    const float* ln1_b  = (const float*)d_in[2];
    const float* w_qkv  = (const float*)d_in[3];
    const float* b_qkv  = (const float*)d_in[4];
    const float* w_proj = (const float*)d_in[5];
    const float* b_proj = (const float*)d_in[6];
    const float* gamma1 = (const float*)d_in[7];
    const float* ln2_g  = (const float*)d_in[8];
    const float* ln2_b  = (const float*)d_in[9];
    const float* w_fc1  = (const float*)d_in[10];
    const float* b_fc1  = (const float*)d_in[11];
    const float* w_fc2  = (const float*)d_in[12];
    const float* b_fc2  = (const float*)d_in[13];
    const float* gamma2 = (const float*)d_in[14];

    char* base = (char*)d_ws;
    size_t off = 0;
    auto alloc = [&](size_t bytes) -> void* {
        void* p = base + off;
        off = (off + bytes + 255) & ~(size_t)255;
        return p;
    };
    const size_t nWqkv = (size_t)3 * Cc * Cc;
    const size_t nWpr  = (size_t)Cc * Cc;
    const size_t nWf1  = (size_t)HIDh * Cc;
    const size_t nWf2  = (size_t)Cc * HIDh;
    const size_t nTok  = (size_t)Mtok * Cc;
    const size_t nHead = (size_t)Bb * Hh * Nn * DHd;   // == nTok
    const size_t nGelu = (size_t)Mtok * HIDh;

    __bf16* wqkv_bf = (__bf16*)alloc(nWqkv * 2);
    __bf16* wproj_bf = (__bf16*)alloc(nWpr * 2);
    __bf16* wfc1_bf = (__bf16*)alloc(nWf1 * 2);
    __bf16* wfc2_bf = (__bf16*)alloc(nWf2 * 2);
    __bf16* h_bf    = (__bf16*)alloc(nTok * 2);
    __bf16* qbuf    = (__bf16*)alloc(nHead * 2);
    __bf16* kbuf    = (__bf16*)alloc(nHead * 2);
    __bf16* vbuf    = (__bf16*)alloc(nHead * 2);
    __bf16* obuf    = (__bf16*)alloc(nTok * 2);
    float*  x1      = (float*)alloc(nTok * 4);
    __bf16* h2_bf   = (__bf16*)alloc(nTok * 2);
    __bf16* gbuf    = (__bf16*)alloc(nGelu * 2);

    // 1) weights -> bf16
    cast_f32_bf16<<<4096, 256, 0, stream>>>(w_qkv,  wqkv_bf, (int)nWqkv);
    cast_f32_bf16<<<4096, 256, 0, stream>>>(w_proj, wproj_bf, (int)nWpr);
    cast_f32_bf16<<<4096, 256, 0, stream>>>(w_fc1,  wfc1_bf, (int)nWf1);
    cast_f32_bf16<<<4096, 256, 0, stream>>>(w_fc2,  wfc2_bf, (int)nWf2);

    // 2) LN1
    layernorm_bf16<<<Mtok, 256, 0, stream>>>(x, ln1_g, ln1_b, h_bf);

    // 3) QKV GEMM  [5480,1024] @ [3072,1024]^T
    gemm_bf16_wmma<0><<<dim3((Mtok + 127) / 128, (3 * Cc) / 128), 256, 0, stream>>>(
        h_bf, wqkv_bf, b_qkv, nullptr, nullptr, nullptr, nullptr,
        qbuf, kbuf, vbuf, Mtok, Cc, 3 * Cc);

    // 4) flash attention
    attn_flash<<<dim3((Nn + 63) / 64, Bb * Hh), 128, 0, stream>>>(qbuf, kbuf, vbuf, obuf);

    // 5) proj GEMM + residual*gamma1 -> x1 (f32)
    gemm_bf16_wmma<1><<<dim3((Mtok + 127) / 128, Cc / 128), 256, 0, stream>>>(
        obuf, wproj_bf, b_proj, x, gamma1, x1, nullptr,
        nullptr, nullptr, nullptr, Mtok, Cc, Cc);

    // 6) LN2
    layernorm_bf16<<<Mtok, 256, 0, stream>>>(x1, ln2_g, ln2_b, h2_bf);

    // 7) fc1 GEMM + exact GELU -> bf16
    gemm_bf16_wmma<2><<<dim3((Mtok + 127) / 128, HIDh / 128), 256, 0, stream>>>(
        h2_bf, wfc1_bf, b_fc1, nullptr, nullptr, nullptr, gbuf,
        nullptr, nullptr, nullptr, Mtok, Cc, HIDh);

    // 8) fc2 GEMM + residual*gamma2 -> d_out (f32)
    gemm_bf16_wmma<1><<<dim3((Mtok + 127) / 128, Cc / 128), 256, 0, stream>>>(
        gbuf, wfc2_bf, b_fc2, x1, gamma2, (float*)d_out, nullptr,
        nullptr, nullptr, nullptr, Mtok, HIDh, Cc);
}
